// TimeAwareMultiHeadAttention_89601607729653
// MI455X (gfx1250) — matrix operand order repaired and checked by reading, not compiled, not run on hardware
//
#include <hip/hip_runtime.h>
#include <hip/hip_bf16.h>

typedef __attribute__((ext_vector_type(16))) _Float16 v16h;
typedef __attribute__((ext_vector_type(8)))  _Float16 v8h;
typedef __attribute__((ext_vector_type(8)))  float    v8f;
typedef __attribute__((ext_vector_type(4)))  float    v4f;

#define BB 2
#define LL 1024
#define DD 64
#define HH 8

// ---- WMMA operand index patterns (CDNA5 ISA §7.12.2, wave32) ----
// A 16x32 f16: lane holds row m = lane%16; element e -> VGPR v=e/2, half s=e%2
//   k = (lane/16)*8 + (v/4)*16 + (v%4)*2 + s
__device__ __forceinline__ int a_kpat(int lane, int e) {
  int v = e >> 1, s = e & 1;
  return ((lane >> 4) << 3) + ((v >> 2) << 4) + ((v & 3) << 1) + s;
}
// inverse: given k (0..31), row m -> (lane, element)
__device__ __forceinline__ void a_inv(int k, int m, int& lane, int& e) {
  int hi = k >> 4;          // v/4
  int g  = k & 15;
  int lh = g >> 3;          // lane half
  int q  = g & 7;           // (v%4)*2 + s
  int v  = hi * 4 + (q >> 1);
  int s  = q & 1;
  lane = lh * 16 + m;
  e = v * 2 + s;
}
// B 32x16 f16: lane holds col n = lane%16; k = (e/8)*16 + (lane/16)*8 + e%8
__device__ __forceinline__ void b_inv(int k, int n, int& lane, int& e) {
  lane = (((k >> 3) & 1) << 4) + n;
  e = ((k >> 4) << 3) + (k & 7);
}
// C/D f32 16x16: element r -> row m = r + (lane/16)*8, col n = lane%16

static __device__ __forceinline__ v8f wmma16(v16h a, v16h b, v8f c) {
  return __builtin_amdgcn_wmma_f32_16x16x32_f16(false, a, false, b, (short)0, c,
                                                false, false);
}

// Workspace tile layouts (all f16):
//  Qa : [bh][it:64][cc:2][lane:32][e:16]   A-operand order (Q/8)
//  Kb : [bh][jt:64][cc:2][lane:32][e:16]   B-operand order (K^T)
//  Vb : [bh][jt32:32][cch:4][lane:32][e:16] B-operand order (V)
//  Pc : [bh][it:64][jt32:32][sub:2][lane:32][r:8]  C-layout order
//  Oa : [mt:128][kc:16][lane:32][e:16]     A-operand order (concat heads)

// =====================================================================
// Kernel 1: Q/K/V projections [2048x64]@[64x512]+bias -> f16, stored
// pre-swizzled into WMMA operand order. grid (128,4,3), block 256.
// =====================================================================
__global__ void proj_kernel(const float* __restrict__ q_in, const float* __restrict__ k_in,
                            const float* __restrict__ v_in,
                            const float* __restrict__ Wq, const float* __restrict__ bq,
                            const float* __restrict__ Wk, const float* __restrict__ bk,
                            const float* __restrict__ Wv, const float* __restrict__ bv,
                            _Float16* __restrict__ Qa, _Float16* __restrict__ Kb,
                            _Float16* __restrict__ Vb) {
  int lane = threadIdx.x & 31;
  int w    = threadIdx.x >> 5;
  int z    = blockIdx.z;
  const float* in = (z == 0) ? q_in : (z == 1) ? k_in : v_in;
  const float* W  = (z == 0) ? Wq   : (z == 1) ? Wk   : Wv;
  const float* bs = (z == 0) ? bq   : (z == 1) ? bk   : bv;
  float scale = (z == 0) ? 0.125f : 1.0f;   // fold 1/sqrt(64) into Q

  int m0 = blockIdx.x * 16;
  int n0 = (blockIdx.y * 8 + w) * 16;
  int mloc = lane & 15;
  int h8   = (lane >> 4) << 3;
  int mrow = m0 + mloc;
  int ncol = n0 + mloc;

  // A operand: 4 runs of 8 consecutive f32 per lane (2x b128 each)
  v16h a0, a1;
  const float* rp = in + (size_t)mrow * DD;
#pragma unroll
  for (int run = 0; run < 4; ++run) {
    v4f lo = *(const v4f*)(rp + h8 + run * 16);
    v4f hi = *(const v4f*)(rp + h8 + run * 16 + 4);
#pragma unroll
    for (int t = 0; t < 4; ++t) {
      if (run < 2) { a0[run * 8 + t] = (_Float16)lo[t]; a0[run * 8 + 4 + t] = (_Float16)hi[t]; }
      else { a1[(run - 2) * 8 + t] = (_Float16)lo[t]; a1[(run - 2) * 8 + 4 + t] = (_Float16)hi[t]; }
    }
  }
  // B operand: column-strided weight gathers (cold kernel, acceptable)
  v16h bm0, bm1;
#pragma unroll
  for (int e = 0; e < 16; ++e) {
    int kb = ((e >> 3) << 4) + ((lane >> 4) << 3) + (e & 7);
    bm0[e] = (_Float16)W[(size_t)kb * 512 + ncol];
    bm1[e] = (_Float16)W[(size_t)(kb + 32) * 512 + ncol];
  }
  v8f c = {};
  c = wmma16(a0, bm0, c);
  c = wmma16(a1, bm1, c);

  float bias = bs[ncol];
  int hh = ncol >> 6, d = ncol & 63;
#pragma unroll
  for (int r = 0; r < 8; ++r) {
    int row = m0 + r + h8;
    int bb = row >> 10, l = row & 1023;
    int bh = bb * HH + hh;
    _Float16 val = (_Float16)((c[r] + bias) * scale);
    if (z == 0) {          // Q -> A-operand order
      int it = l >> 4, m = l & 15;
      int cc = d >> 5, kk = d & 31;
      int lt, et; a_inv(kk, m, lt, et);
      Qa[(size_t)((bh * 64 + it) * 2 + cc) * 512 + lt * 16 + et] = val;
    } else if (z == 1) {   // K -> B-operand order (K^T)
      int jt = l >> 4, n = l & 15;
      int cc = d >> 5, kk = d & 31;
      int lt, et; b_inv(kk, n, lt, et);
      Kb[(size_t)((bh * 64 + jt) * 2 + cc) * 512 + lt * 16 + et] = val;
    } else {               // V -> B-operand order
      int jt32 = l >> 5, kkj = l & 31;
      int cch = d >> 4, n = d & 15;
      int lt, et; b_inv(kkj, n, lt, et);
      Vb[(size_t)((bh * 32 + jt32) * 4 + cch) * 512 + lt * 16 + et] = val;
    }
  }
}

// =====================================================================
// Kernel 2: positional scores. F[j,d]=|pos_i - pos_j| (16x64 A operand)
// @ Wp (B operand, 8 valid cols) -> P[j,h], +bp, stored C-layout order.
// grid 16384, block 256; wave -> (b, i, 16-wide j-tile)
// =====================================================================
__global__ void pscore_kernel(const float* __restrict__ pos, const float* __restrict__ Wp,
                              const float* __restrict__ bp, _Float16* __restrict__ Pc) {
  int lane = threadIdx.x & 31;
  int w    = threadIdx.x >> 5;
  int wid  = blockIdx.x * 8 + w;        // < 131072
  int jt = wid & 63;
  int i  = (wid >> 6) & 1023;
  int b  = wid >> 16;
  int j0 = jt * 16;
  int mloc = lane & 15;
  int h8   = (lane >> 4) << 3;
  int jr = j0 + mloc;                   // F row = j'
  const float* pi = pos + (size_t)(b * LL + i) * DD;
  const float* pj = pos + (size_t)(b * LL + jr) * DD;
  int n = mloc;                         // col = head

  v16h a0, a1;
#pragma unroll
  for (int run = 0; run < 4; ++run) {
    int off = h8 + run * 16;
    v4f il = *(const v4f*)(pi + off);
    v4f ih = *(const v4f*)(pi + off + 4);
    v4f jl = *(const v4f*)(pj + off);
    v4f jh = *(const v4f*)(pj + off + 4);
#pragma unroll
    for (int t = 0; t < 4; ++t) {
      _Float16 x0 = (_Float16)fabsf(il[t] - jl[t]);
      _Float16 x1 = (_Float16)fabsf(ih[t] - jh[t]);
      if (run < 2) { a0[run * 8 + t] = x0; a0[run * 8 + 4 + t] = x1; }
      else { a1[(run - 2) * 8 + t] = x0; a1[(run - 2) * 8 + 4 + t] = x1; }
    }
  }
  v16h bm0, bm1;
#pragma unroll
  for (int e = 0; e < 16; ++e) {
    int kb = ((e >> 3) << 4) + ((lane >> 4) << 3) + (e & 7);
    bm0[e] = (n < HH) ? (_Float16)Wp[kb * HH + n] : (_Float16)0.0f;
    bm1[e] = (n < HH) ? (_Float16)Wp[(kb + 32) * HH + n] : (_Float16)0.0f;
  }
  v8f c = {};
  c = wmma16(a0, bm0, c);
  c = wmma16(a1, bm1, c);

  if (n < HH) {
    float bbp = bp[n];
    int bh = b * HH + n;
    int it = i >> 4, ri = i & 15;
#pragma unroll
    for (int r = 0; r < 8; ++r) {
      int j = j0 + r + h8;
      int jt32 = j >> 5, sub = (j >> 4) & 1, nj = j & 15;
      int lt = ((ri >> 3) << 4) + nj;
      int rr = ri & 7;
      Pc[(size_t)((((bh * 64 + it) * 32 + jt32) * 2 + sub) * 32 + lt) * 8 + rr] =
          (_Float16)(c[r] + bbp);
    }
  }
}

// =====================================================================
// Kernel 3: fused flash attention; all operands pre-swizzled -> inner
// loop is pure contiguous b128 loads + 8 WMMAs per 32 columns.
// grid 128, block 256 (8 waves; wave -> (b,h,16-row i-tile))
// =====================================================================
__global__ void attn_kernel(const _Float16* __restrict__ Qa, const _Float16* __restrict__ Kb,
                            const _Float16* __restrict__ Vb, const _Float16* __restrict__ Pc,
                            _Float16* __restrict__ Oa) {
  __shared__ _Float16 pls[8][16][32];   // per-wave prob tile (8 KB)
  int lane = threadIdx.x & 31;
  int w    = threadIdx.x >> 5;
  int wid  = blockIdx.x * 8 + w;        // < 1024
  int it = wid & 63;
  int h  = (wid >> 6) & 7;
  int b  = wid >> 9;
  int i0 = it * 16;
  int bh = b * HH + h;
  int mloc = lane & 15;
  int h8   = (lane >> 4) << 3;

  v16h qa0 = *(const v16h*)(Qa + (size_t)((bh * 64 + it) * 2 + 0) * 512 + lane * 16);
  v16h qa1 = *(const v16h*)(Qa + (size_t)((bh * 64 + it) * 2 + 1) * 512 + lane * 16);

  float mrow[8], lrow[8];
  v8f acc[4] = {{}, {}, {}, {}};
#pragma unroll
  for (int r = 0; r < 8; ++r) { mrow[r] = -3.0e38f; lrow[r] = 0.0f; }

  for (int jt32 = 0; jt32 < 32; ++jt32) {
    int jt0 = jt32 * 2, jt1 = jt32 * 2 + 1;
    // ---- S = Q K^T : contiguous B-operand tiles ----
    v16h kb00 = *(const v16h*)(Kb + (size_t)((bh * 64 + jt0) * 2 + 0) * 512 + lane * 16);
    v16h kb01 = *(const v16h*)(Kb + (size_t)((bh * 64 + jt0) * 2 + 1) * 512 + lane * 16);
    v16h kb10 = *(const v16h*)(Kb + (size_t)((bh * 64 + jt1) * 2 + 0) * 512 + lane * 16);
    v16h kb11 = *(const v16h*)(Kb + (size_t)((bh * 64 + jt1) * 2 + 1) * 512 + lane * 16);
    v8f zz = {};
    v8f s0 = wmma16(qa0, kb00, zz); s0 = wmma16(qa1, kb01, s0);
    v8f s1 = wmma16(qa0, kb10, zz); s1 = wmma16(qa1, kb11, s1);
    // ---- positional scores: C-layout order, one b128 per subtile ----
    const _Float16* pbase = Pc + (size_t)(((bh * 64 + it) * 32 + jt32) * 2) * 256;
    v8h pv0 = *(const v8h*)(pbase + (size_t)lane * 8);
    v8h pv1 = *(const v8h*)(pbase + 256 + (size_t)lane * 8);
    // ---- online softmax over 32 cols ----
    float p0[8], p1[8], fac[8];
#pragma unroll
    for (int r = 0; r < 8; ++r) {
      float t0 = s0[r] + (float)pv0[r];
      float t1 = s1[r] + (float)pv1[r];
      float rm = fmaxf(t0, t1);
      rm = fmaxf(rm, __shfl_xor(rm, 1, 32));
      rm = fmaxf(rm, __shfl_xor(rm, 2, 32));
      rm = fmaxf(rm, __shfl_xor(rm, 4, 32));
      rm = fmaxf(rm, __shfl_xor(rm, 8, 32));
      float nm = fmaxf(mrow[r], rm);
      fac[r] = __expf(mrow[r] - nm);
      mrow[r] = nm;
      p0[r] = __expf(t0 - nm);
      p1[r] = __expf(t1 - nm);
      float rs = p0[r] + p1[r];
      rs += __shfl_xor(rs, 1, 32);
      rs += __shfl_xor(rs, 2, 32);
      rs += __shfl_xor(rs, 4, 32);
      rs += __shfl_xor(rs, 8, 32);
      lrow[r] = lrow[r] * fac[r] + rs;
#pragma unroll
      for (int cch = 0; cch < 4; ++cch) acc[cch][r] *= fac[r];
    }
    // ---- probs: C-layout -> LDS -> A-layout (16x32 f16) ----
    __syncthreads();
#pragma unroll
    for (int r = 0; r < 8; ++r) {
      pls[w][r + h8][mloc]      = (_Float16)p0[r];
      pls[w][r + h8][16 + mloc] = (_Float16)p1[r];
    }
    __syncthreads();
    v16h pa;
#pragma unroll
    for (int e = 0; e < 16; ++e)
      pa[e] = pls[w][mloc][a_kpat(lane, e)];
    // ---- O += probs @ V (K=32 fully used) ----
#pragma unroll
    for (int cch = 0; cch < 4; ++cch) {
      v16h vb = *(const v16h*)(Vb + (size_t)((bh * 32 + jt32) * 4 + cch) * 512 + lane * 16);
      acc[cch] = wmma16(pa, vb, acc[cch]);
    }
  }
  // ---- normalize; store concat-head result in A-operand order ----
#pragma unroll
  for (int r = 0; r < 8; ++r) {
    int irow = i0 + r + h8;
    float inv = 1.0f / lrow[r];
    int row_g = b * LL + irow;
    int mt = row_g >> 4, m = row_g & 15;
#pragma unroll
    for (int cch = 0; cch < 4; ++cch) {
      int col = h * 64 + cch * 16 + mloc;
      int kc = col >> 5, kk = col & 31;
      int lt, et; a_inv(kk, m, lt, et);
      Oa[(size_t)(mt * 16 + kc) * 512 + lt * 16 + et] = (_Float16)(acc[cch][r] * inv);
    }
  }
}

// =====================================================================
// Kernel 4: output projection [2048x512]@[512x64]+bo -> f32 out.
// A operand loads are contiguous b128 (pre-swizzled Oa).
// grid 64, block 256 (8 waves; wave -> one 16x16 tile, 16 WMMAs)
// =====================================================================
__global__ void outproj_kernel(const _Float16* __restrict__ Oa, const float* __restrict__ Wo,
                               const float* __restrict__ bo, float* __restrict__ out) {
  int lane = threadIdx.x & 31;
  int w    = threadIdx.x >> 5;
  int wid  = blockIdx.x * 8 + w;        // < 512
  int nt = wid & 3;
  int mt = wid >> 2;
  int m0 = mt * 16, n0 = nt * 16;
  int mloc = lane & 15;

  v8f c = {};
  for (int kc = 0; kc < 16; ++kc) {
    v16h a = *(const v16h*)(Oa + (size_t)(mt * 16 + kc) * 512 + lane * 16);
    v16h bm;
#pragma unroll
    for (int e = 0; e < 16; ++e) {
      int kb = kc * 32 + ((e >> 3) << 4) + ((lane >> 4) << 3) + (e & 7);
      bm[e] = (_Float16)Wo[(size_t)kb * 64 + n0 + mloc];
    }
    c = wmma16(a, bm, c);
  }
  float bias = bo[n0 + mloc];
#pragma unroll
  for (int r = 0; r < 8; ++r) {
    int row = m0 + r + ((lane >> 4) << 3);
    out[(size_t)row * 64 + n0 + mloc] = c[r] + bias;
  }
}

extern "C" void kernel_launch(void* const* d_in, const int* in_sizes, int n_in,
                              void* d_out, int out_size, void* d_ws, size_t ws_size,
                              hipStream_t stream) {
  (void)in_sizes; (void)n_in; (void)out_size; (void)ws_size;
  const float* query = (const float*)d_in[0];
  const float* key   = (const float*)d_in[1];
  const float* value = (const float*)d_in[2];
  const float* pos   = (const float*)d_in[3];
  const float* Wq = (const float*)d_in[4];
  const float* bq = (const float*)d_in[5];
  const float* Wk = (const float*)d_in[6];
  const float* bk = (const float*)d_in[7];
  const float* Wv = (const float*)d_in[8];
  const float* bv = (const float*)d_in[9];
  const float* Wp = (const float*)d_in[10];
  const float* bp = (const float*)d_in[11];
  const float* Wo = (const float*)d_in[12];
  const float* bo = (const float*)d_in[13];
  float* out = (float*)d_out;

  // workspace (f16 elems): Qa | Kb | Vb | Pc | Oa  (~40 MB)
  _Float16* Qa = (_Float16*)d_ws;
  _Float16* Kb = Qa + (size_t)BB * HH * LL * DD;     // +1,048,576
  _Float16* Vb = Kb + (size_t)BB * HH * LL * DD;
  _Float16* Pc = Vb + (size_t)BB * HH * LL * DD;
  _Float16* Oa = Pc + (size_t)BB * HH * LL * LL;     // +16,777,216

  proj_kernel<<<dim3(128, 4, 3), 256, 0, stream>>>(query, key, value,
                                                   Wq, bq, Wk, bk, Wv, bv,
                                                   Qa, Kb, Vb);
  pscore_kernel<<<dim3(16384), 256, 0, stream>>>(pos, Wp, bp, Pc);
  attn_kernel<<<dim3(128), 256, 0, stream>>>(Qa, Kb, Vb, Pc, Oa);
  outproj_kernel<<<dim3(64), 256, 0, stream>>>(Oa, Wo, bo, out);
}
